// img_attention_model_44865228374113
// MI455X (gfx1250) — matrix operand order repaired and checked by reading, not compiled, not run on hardware
//
#include <hip/hip_runtime.h>
#include <hip/hip_bf16.h>
#include <math.h>

// Dims from reference
#define B_SZ 256
#define N_SZ 196
#define F_SZ 512
#define E_SZ 512
#define H_SZ 1024
#define NT   13            // ceil(196/16) n-tiles
#define NPAD (NT * 16)     // 208

// CDNA5 async global->LDS copy path (GLOBAL_LOAD_ASYNC_TO_LDS_B128, ASYNCcnt).
// Set to 0 to fall back to regs->ds_store staging if the asm fails to parse.
#define USE_ASYNC_COPY 1

typedef __attribute__((ext_vector_type(16))) __bf16 v16bf;
typedef __attribute__((ext_vector_type(8)))  float  v8f;

#if __has_builtin(__builtin_amdgcn_tanhf)
__device__ __forceinline__ float fast_tanhf(float x) { return __builtin_amdgcn_tanhf(x); }
#else
__device__ __forceinline__ float fast_tanhf(float x) { return tanhf(x); }
#endif

__device__ __forceinline__ unsigned short f32_to_bf16(float f) {
  union { float f; unsigned int u; } c; c.f = f;
  unsigned int u = c.u;
  // round-to-nearest-even
  return (unsigned short)((u + 0x7FFFu + ((u >> 16) & 1u)) >> 16);
}

#if USE_ASYNC_COPY
// 16-byte async copy: global (sgpr base + vgpr byte offset) -> LDS byte address.
// LDS byte address = low 32 bits of the generic shared pointer (ISA 10.2:
// LDS aperture puts the LDS offset in addr[31:0]).
__device__ __forceinline__ void async_copy16(const void* gbase, unsigned goff,
                                             const void* lds) {
  unsigned lds_addr = (unsigned)(unsigned long long)lds;
  asm volatile("global_load_async_to_lds_b128 %0, %1, %2"
               :
               : "v"(lds_addr), "v"(goff), "s"(gbase)
               : "memory");
}
__device__ __forceinline__ void async_wait0() {
  asm volatile("s_wait_asynccnt 0x0" ::: "memory");
}
#endif

// Inverse of the 16-bit A-fragment layout (contiguous K order):
// lane = m + 16*h; within a 32-wide K step, lane-half h covers
// k_local = 16h..16h+15 (i=0..15).
// Any 4-aligned K-quad maps to one lane, consecutive i -> ushort4-packable.
__device__ __forceinline__ int swz_h(int kk) { return (kk >> 4) & 1; }
__device__ __forceinline__ int swz_i(int kk) { return kk & 15; }

// ---------------------------------------------------------------------------
// One-time: dst[e*K + k] = bf16(src[k*E + e])  (coalesced reads; tiny kernel)
// ---------------------------------------------------------------------------
__global__ __launch_bounds__(256)
void transpose_bf16_kernel(const float* __restrict__ src,
                           unsigned short* __restrict__ dst, int K, int E) {
  int idx = blockIdx.x * 256 + threadIdx.x;
  if (idx < K * E) {
    int k = idx / E, e = idx % E;
    dst[(size_t)e * K + k] = f32_to_bf16(src[idx]);
  }
}

// ---------------------------------------------------------------------------
// Stage A: x2[b,e] = tanh(hidden[b,:] @ W2 + b2[e])   [256 x 512], via WMMA
// grid = (16 b-tiles, 32 e-tiles), block = 32 (one wave)
// ---------------------------------------------------------------------------
__global__ __launch_bounds__(32)
void x2_gemm_kernel(const float* __restrict__ hidden,        // [1,B,H]
                    const unsigned short* __restrict__ W2T,  // [E,H] bf16
                    const float* __restrict__ b2,            // [E]
                    float* __restrict__ x2out) {             // [B,E]
  __shared__ __align__(32) unsigned short hswz[32 * 512];   // 16 x 1024, A-swizzled
  __shared__ __align__(32) unsigned short wt2[16 * 1040];   // wt2[n][k] = W2[k][e0+n]

  const int tid = threadIdx.x;
  const int b0  = blockIdx.x * 16;
  const int e0  = blockIdx.y * 16;

  // Stage W2 tile from pre-transposed bf16 (pure copy -> async to LDS)
#if USE_ASYNC_COPY
  for (int idx = tid; idx < 16 * H_SZ / 8; idx += 32) {
    int n = idx >> 7;                 // 128 octets per row
    int k = (idx & 127) << 3;
    unsigned goff = (unsigned)((((e0 + n) << 10) + k) * sizeof(unsigned short));
    async_copy16(W2T, goff, &wt2[n * 1040 + k]);
  }
#else
  for (int idx = tid; idx < 16 * H_SZ / 8; idx += 32) {
    int n = idx >> 7;
    int k = (idx & 127) << 3;
    *(uint4*)&wt2[n * 1040 + k] = *(const uint4*)&W2T[((size_t)(e0 + n) << 10) + k];
  }
#endif
  // Stage hidden tile: float4 coalesced reads -> ushort4 packed swizzled stores
  for (int idx = tid; idx < 16 * H_SZ / 4; idx += 32) {
    int m = idx >> 8;                 // 256 quads per row
    int k = (idx & 255) << 2;
    float4 v4 = *(const float4*)&hidden[((size_t)(b0 + m) << 10) + k];
    int s = k >> 5, kk = k & 31;
    int ln = m + (swz_h(kk) << 4);
    int base = s * 512 + ln * 16 + swz_i(kk);
    ushort4 p;
    p.x = f32_to_bf16(v4.x); p.y = f32_to_bf16(v4.y);
    p.z = f32_to_bf16(v4.z); p.w = f32_to_bf16(v4.w);
    *(ushort4*)&hswz[base] = p;
  }
#if USE_ASYNC_COPY
  async_wait0();
#endif
  __syncthreads();

  const int lane = tid & 31;
  const int col  = lane & 15;
  const int h    = lane >> 4;

  v8f acc = {};
  #pragma unroll
  for (int s = 0; s < 32; ++s) {
    v16bf a = *(const v16bf*)&hswz[s * 512 + lane * 16];
    v16bf b = *(const v16bf*)&wt2[col * 1040 + 32 * s + 16 * h];
    acc = __builtin_amdgcn_wmma_f32_16x16x32_bf16(false, a, false, b,
                                                  (short)0, acc, false, false);
  }
  const float bias = b2[e0 + col];
  #pragma unroll
  for (int r = 0; r < 8; ++r) {
    int m = r + 8 * h;                // C layout: VGPR r -> row r (+8 for hi half)
    x2out[(size_t)(b0 + m) * E_SZ + e0 + col] = fast_tanhf(acc[r] + bias);
  }
}

// ---------------------------------------------------------------------------
// Stage B: per-batch fused  tanh(x@W1+b1) . x2  -> softmax -> attn-weighted sum
// grid = 256 (one block per batch), block = 416 (13 waves, one wave per n-tile)
// ---------------------------------------------------------------------------
__global__ __launch_bounds__(416)
void attn_kernel(const float* __restrict__ x,               // [B,N,F]
                 const unsigned short* __restrict__ W1T,    // [E,F] bf16
                 const float* __restrict__ b1,               // [E]
                 const float* __restrict__ x2,               // [B,E]
                 float* __restrict__ out) {                  // [B,F]
  __shared__ __align__(32) unsigned short xswz[NT * 16 * 512]; // 208 KB: x[b] bf16 A-swizzled
  __shared__ __align__(32) unsigned short wtbuf[2][16 * 528];  // double-buffered W1 e-tile
  __shared__ float x2s[E_SZ];
  __shared__ float b1s[E_SZ];
  __shared__ float scoresS[NPAD];
  __shared__ float attnS[NPAD];

  const int tid  = threadIdx.x;       // 0..415
  const int b    = blockIdx.x;
  const int lane = tid & 31;
  const int w    = tid >> 5;          // wave id == n-tile index (0..12)
  const int col  = lane & 15;
  const int h    = lane >> 4;
  const float* xb = x + (size_t)b * N_SZ * F_SZ;

  for (int e = tid; e < E_SZ; e += 416) {
    x2s[e] = x2[(size_t)b * E_SZ + e];
    b1s[e] = b1[e];
  }
  // Stage x[b]: float4 coalesced reads -> ushort4 packed swizzled LDS stores
  for (int idx = tid; idx < NPAD * F_SZ / 4; idx += 416) {
    int n = idx >> 7;                 // 128 quads per row
    int k = (idx & 127) << 2;
    float4 v4 = (n < N_SZ) ? *(const float4*)&xb[((size_t)n << 9) + k]
                           : make_float4(0.f, 0.f, 0.f, 0.f);
    int t = n >> 4, m = n & 15;
    int s = k >> 5, kk = k & 31;
    int ln = m + (swz_h(kk) << 4);
    int base = t * 8192 + s * 512 + ln * 16 + swz_i(kk);
    ushort4 p;
    p.x = f32_to_bf16(v4.x); p.y = f32_to_bf16(v4.y);
    p.z = f32_to_bf16(v4.z); p.w = f32_to_bf16(v4.w);
    *(ushort4*)&xswz[base] = p;
  }

  // W1 e-tile staging: 1024 x 16B per tile, <=3 chunks per thread.
#if USE_ASYNC_COPY
  #define STAGE_W1(e0_, buf_)                                               \
    _Pragma("unroll")                                                       \
    for (int j = 0; j < 3; ++j) {                                           \
      int idx = tid + j * 416;                                              \
      if (idx < 1024) {                                                     \
        int n = idx >> 6;                                                   \
        int k = (idx & 63) << 3;                                            \
        unsigned goff = (unsigned)(((((e0_) + n) << 9) + k) * 2u);          \
        async_copy16(W1T, goff, &(buf_)[n * 528 + k]);                      \
      }                                                                     \
    }
  STAGE_W1(0, wtbuf[0]);
  async_wait0();
#else
  uint4 sreg[3];
  #define ISSUE_W1(e0_)                                                     \
    _Pragma("unroll")                                                       \
    for (int j = 0; j < 3; ++j) {                                           \
      int idx = tid + j * 416;                                              \
      if (idx < 1024) {                                                     \
        int n = idx >> 6;                                                   \
        int k = (idx & 63) << 3;                                            \
        sreg[j] = *(const uint4*)&W1T[((size_t)((e0_) + n) << 9) + k];      \
      }                                                                     \
    }
  #define COMMIT_W1(buf_)                                                   \
    _Pragma("unroll")                                                       \
    for (int j = 0; j < 3; ++j) {                                           \
      int idx = tid + j * 416;                                              \
      if (idx < 1024) {                                                     \
        int n = idx >> 6;                                                   \
        int k = (idx & 63) << 3;                                            \
        *(uint4*)&(buf_)[n * 528 + k] = sreg[j];                            \
      }                                                                     \
    }
  ISSUE_W1(0);
  COMMIT_W1(wtbuf[0]);
#endif
  __syncthreads();

  // Each wave keeps its n-tile's 16 A-fragments in registers (reused x32 e-tiles)
  v16bf afrag[16];
  {
    const unsigned short* base = &xswz[w * 8192 + lane * 16];
    #pragma unroll
    for (int s = 0; s < 16; ++s) afrag[s] = *(const v16bf*)(base + s * 512);
  }

  float sacc[8] = {0.f, 0.f, 0.f, 0.f, 0.f, 0.f, 0.f, 0.f};

  for (int et = 0; et < 32; ++et) {
    const int e0 = et * 16;
    const unsigned short* cur = wtbuf[et & 1];
    unsigned short* nxt = wtbuf[(et & 1) ^ 1];

    // Prefetch next tile: async copies proceed in background during the WMMAs
#if USE_ASYNC_COPY
    if (et + 1 < 32) { STAGE_W1(e0 + 16, nxt); }
#else
    if (et + 1 < 32) { ISSUE_W1(e0 + 16); }
#endif

    v8f acc = {};
    #pragma unroll
    for (int s = 0; s < 16; ++s) {
      v16bf bf = *(const v16bf*)&cur[col * 528 + 32 * s + 16 * h];
      acc = __builtin_amdgcn_wmma_f32_16x16x32_bf16(false, afrag[s], false, bf,
                                                    (short)0, acc, false, false);
    }

#if !USE_ASYNC_COPY
    if (et + 1 < 32) { COMMIT_W1(nxt); }
#endif

    const float bias  = b1s[e0 + col];
    const float scale = x2s[e0 + col];
    #pragma unroll
    for (int r = 0; r < 8; ++r)
      sacc[r] += fast_tanhf(acc[r] + bias) * scale;   // partial scores over e

#if USE_ASYNC_COPY
    async_wait0();     // drain this wave's async copies into nxt
#endif
    __syncthreads();   // all waves staged; cur free for restage next iter
  }

  // Reduce each row's partial score across the 16 lanes of each half-wave
  #pragma unroll
  for (int r = 0; r < 8; ++r) {
    float v = sacc[r];
    v += __shfl_xor(v, 1, 32);
    v += __shfl_xor(v, 2, 32);
    v += __shfl_xor(v, 4, 32);
    v += __shfl_xor(v, 8, 32);
    sacc[r] = v;
  }
  if (lane == 0 || lane == 16) {
    int base = w * 16 + h * 8;   // lane0 -> rows 0..7, lane16 -> rows 8..15
    #pragma unroll
    for (int r = 0; r < 8; ++r) {
      int n = base + r;
      scoresS[n] = (n < N_SZ) ? sacc[r] : -1e30f;
    }
  }
  __syncthreads();

  // Softmax over N=196 by wave 0
  if (tid < 32) {
    float m = -1e30f;
    for (int n = lane; n < N_SZ; n += 32) m = fmaxf(m, scoresS[n]);
    m = fmaxf(m, __shfl_xor(m, 1, 32));
    m = fmaxf(m, __shfl_xor(m, 2, 32));
    m = fmaxf(m, __shfl_xor(m, 4, 32));
    m = fmaxf(m, __shfl_xor(m, 8, 32));
    m = fmaxf(m, __shfl_xor(m, 16, 32));
    float ssum = 0.f;
    for (int n = lane; n < N_SZ; n += 32) {
      float e = __expf(scoresS[n] - m);
      attnS[n] = e;
      ssum += e;
    }
    ssum += __shfl_xor(ssum, 1, 32);
    ssum += __shfl_xor(ssum, 2, 32);
    ssum += __shfl_xor(ssum, 4, 32);
    ssum += __shfl_xor(ssum, 8, 32);
    ssum += __shfl_xor(ssum, 16, 32);
    float inv = 1.0f / ssum;
    for (int n = lane; n < N_SZ; n += 32) attnS[n] *= inv;
  }
  __syncthreads();

  // out[b,f] = sum_n attn[n] * x[b,n,f]  (fp32 x re-read; L2-resident)
  for (int f = tid; f < F_SZ; f += 416) {
    float acc = 0.f;
    const float* xp = xb + f;
    for (int n = 0; n < N_SZ; ++n) acc += attnS[n] * xp[(size_t)n * F_SZ];
    out[(size_t)b * F_SZ + f] = acc;
  }
}

extern "C" void kernel_launch(void* const* d_in, const int* in_sizes, int n_in,
                              void* d_out, int out_size, void* d_ws, size_t ws_size,
                              hipStream_t stream) {
  const float* x      = (const float*)d_in[0];
  const float* hidden = (const float*)d_in[1];
  const float* W1     = (const float*)d_in[2];
  const float* b1     = (const float*)d_in[3];
  const float* W2     = (const float*)d_in[4];
  const float* b2     = (const float*)d_in[5];
  float* out = (float*)d_out;

  // Workspace layout: x2 fp32 (512 KB) | W1T bf16 (512 KB) | W2T bf16 (1 MB)
  float*          x2  = (float*)d_ws;
  unsigned short* W1T = (unsigned short*)((char*)d_ws + (512u << 10));
  unsigned short* W2T = (unsigned short*)((char*)d_ws + (1024u << 10));

  transpose_bf16_kernel<<<(F_SZ * E_SZ) / 256, 256, 0, stream>>>(W1, W1T, F_SZ, E_SZ);
  transpose_bf16_kernel<<<(H_SZ * E_SZ) / 256, 256, 0, stream>>>(W2, W2T, H_SZ, E_SZ);

  dim3 gA(B_SZ / 16, E_SZ / 16);             // 16 x 32 one-wave blocks
  x2_gemm_kernel<<<gA, 32, 0, stream>>>(hidden, W2T, b2, x2);

  attn_kernel<<<B_SZ, 416, 0, stream>>>(x, W1T, b1, x2, out);
}
